// LaBSE_Switch_9225589752221
// MI455X (gfx1250) — compile-verified, matrix operand order checked by reading, not run on hardware
//
#include <hip/hip_runtime.h>
#include <hip/hip_bf16.h>

#define D_MODEL   768
#define HEADS     12
#define DK        64
#define D_FF      3072
#define N_EXPERTS 8
#define N_LAYERS  4
#define BATCH     32
#define SEQ       128
#define T_TOK     (BATCH * SEQ)          // 4096
#define CAPACITY  (T_TOK / N_EXPERTS)    // 512
#define LN_EPS    1e-5f

typedef __bf16 bf16_t;
typedef __attribute__((ext_vector_type(16))) __bf16 v16bf;
typedef __attribute__((ext_vector_type(8)))  __bf16 v8bf;
typedef __attribute__((ext_vector_type(8)))  float  v8f;

#define LDK 40   // LDS row stride (bf16 elems): 80B = 5*16B, keeps b128 aligned

// ---------------------------------------------------------------------------
// Load one 16x32 bf16 WMMA fragment from an LDS row (row stride = LDK elems).
// ISA layout (16-bit A/B 16x32): lanes 0-15 hold K=0..7 (vgpr0-3) and
// K=16..23 (vgpr4-7); lanes 16-31 hold K=8..15 and K=24..31.  Each half is a
// contiguous 8xbf16 chunk -> one ds_load_b128 each.
// ---------------------------------------------------------------------------
__device__ __forceinline__ v16bf frag_ld(const bf16_t* row, int half) {
    v8bf lo = *(const v8bf*)(row + half * 8);
    v8bf hi = *(const v8bf*)(row + 16 + half * 8);
    v16bf r;
#pragma unroll
    for (int i = 0; i < 8; ++i) { r[i] = lo[i]; r[i + 8] = hi[i]; }
    return r;
}

// ---------------------------------------------------------------------------
// Generic batched GEMM: C[z] = A[z] (MxK) * B[z] (KxN, or NxK if transB) + bias
// Block = 256 threads (8 wave32), tile 128x128, K-step 32, double-buffered LDS.
// Wave (wm,wn) owns a 32x64 sub-tile: 2 A-frags x 4 B-frags -> 8 WMMA / K-step.
// M must be a multiple of 128, K a multiple of 32; N guarded (handles N=64).
// bhA/bhB/bhC: add attention offset (z/HEADS)*D_MODEL + (z%HEADS)*DK to base.
// ---------------------------------------------------------------------------
__global__ __launch_bounds__(256)
void gemm_bf16_wmma(const float* __restrict__ A, const float* __restrict__ B,
                    const float* __restrict__ bias, float* __restrict__ C,
                    int M, int N, int K, int lda, int ldb, int ldc,
                    long long sA, long long sB, long long sC, long long sBias,
                    int bhA, int bhB, int bhC, int transB, int relu)
{
    __shared__ bf16_t As[2][128][LDK];   // [buf][m][k]
    __shared__ bf16_t Bs[2][128][LDK];   // [buf][n][k]  (B staged transposed)

    const int tid = threadIdx.x;
    const int z   = blockIdx.z;
    const long long bh = (long long)(z / HEADS) * D_MODEL + (long long)(z % HEADS) * DK;

    const float* Ab = A + z * sA + (bhA ? bh : 0);
    const float* Bb = B + z * sB + (bhB ? bh : 0);
    float*       Cb = C + z * sC + (bhC ? bh : 0);
    const float* biasb = bias ? (bias + z * sBias) : nullptr;

    const int bm = blockIdx.y * 128;
    const int bn = blockIdx.x * 128;

    const int lane = tid & 31;
    const int wv   = tid >> 5;           // wave id 0..7
    const int wm   = wv & 3;             // M stripe (32 rows)
    const int wn   = wv >> 2;            // N stripe (64 cols)
    const int m16  = lane & 15;
    const int half = lane >> 4;

    // staging coordinates (256 threads, 16 elems each)
    const int arow = tid >> 1;                 // A: 128 rows x 32 k
    const int acb  = (tid & 1) * 16;
    const int bkr  = tid >> 3;                 // B(KxN): 32 k-rows x 128 n
    const int bnb  = (tid & 7) * 16;

    v8f acc[2][4] = {};

    auto stage = [&](int p, int kk) {
        // ---- A tile 128x32 (fp32 -> bf16) ---------------------------------
        {
            const float* src = Ab + (long long)(bm + arow) * lda + (kk + acb);
            if (kk + 32 < K) __builtin_prefetch(src + 32, 0, 1);
#pragma unroll
            for (int j = 0; j < 16; j += 4) {
                float4 f = *(const float4*)(src + j);
                As[p][arow][acb + j + 0] = (bf16_t)f.x;
                As[p][arow][acb + j + 1] = (bf16_t)f.y;
                As[p][arow][acb + j + 2] = (bf16_t)f.z;
                As[p][arow][acb + j + 3] = (bf16_t)f.w;
            }
        }
        // ---- B tile -> Bs[n][k] -------------------------------------------
        if (transB) {   // B is NxK row-major: rows already K-contiguous
            if (bn + arow < N) {
                const float* src = Bb + (long long)(bn + arow) * ldb + (kk + acb);
                if (kk + 32 < K) __builtin_prefetch(src + 32, 0, 1);
#pragma unroll
                for (int j = 0; j < 16; j += 4) {
                    float4 f = *(const float4*)(src + j);
                    Bs[p][arow][acb + j + 0] = (bf16_t)f.x;
                    Bs[p][arow][acb + j + 1] = (bf16_t)f.y;
                    Bs[p][arow][acb + j + 2] = (bf16_t)f.z;
                    Bs[p][arow][acb + j + 3] = (bf16_t)f.w;
                }
            } else {
#pragma unroll
                for (int j = 0; j < 16; ++j) Bs[p][arow][acb + j] = (bf16_t)0.0f;
            }
        } else {        // B is KxN row-major: transpose while staging
            if (bn + bnb < N) {
                const float* src = Bb + (long long)(kk + bkr) * ldb + (bn + bnb);
                if (kk + 32 < K) __builtin_prefetch(src + (long long)32 * ldb, 0, 1);
#pragma unroll
                for (int j = 0; j < 16; j += 4) {
                    float4 f = *(const float4*)(src + j);
                    Bs[p][bnb + j + 0][bkr] = (bf16_t)f.x;
                    Bs[p][bnb + j + 1][bkr] = (bf16_t)f.y;
                    Bs[p][bnb + j + 2][bkr] = (bf16_t)f.z;
                    Bs[p][bnb + j + 3][bkr] = (bf16_t)f.w;
                }
            } else {
#pragma unroll
                for (int j = 0; j < 16; ++j) Bs[p][bnb + j][bkr] = (bf16_t)0.0f;
            }
        }
    };

    stage(0, 0);
    __syncthreads();

    const int nsteps = K / 32;
    for (int s = 0; s < nsteps; ++s) {
        const int p = s & 1;
        if (s + 1 < nsteps) stage(p ^ 1, (s + 1) * 32);   // overlap with compute

        v16bf a0 = frag_ld(&As[p][wm * 32 + m16][0], half);
        v16bf a1 = frag_ld(&As[p][wm * 32 + 16 + m16][0], half);
#pragma unroll
        for (int j = 0; j < 4; ++j) {
            v16bf bf = frag_ld(&Bs[p][wn * 64 + j * 16 + m16][0], half);
            acc[0][j] = __builtin_amdgcn_wmma_f32_16x16x32_bf16(
                false, a0, false, bf, (short)0, acc[0][j], false, false);
            acc[1][j] = __builtin_amdgcn_wmma_f32_16x16x32_bf16(
                false, a1, false, bf, (short)0, acc[1][j], false, false);
        }
        __syncthreads();
    }

    // ---- epilogue: C/D layout = vgpr r: lanes0-15 M=r, lanes16-31 M=8+r ----
#pragma unroll
    for (int j = 0; j < 4; ++j) {
        const int col = bn + wn * 64 + j * 16 + m16;
        if (col >= N) continue;                       // wave-uniform for N=64
        const float bval = biasb ? biasb[col] : 0.0f;
#pragma unroll
        for (int i = 0; i < 2; ++i) {
#pragma unroll
            for (int r = 0; r < 8; ++r) {
                const int mrow = bm + wm * 32 + i * 16 + half * 8 + r;
                float v = acc[i][j][r] + bval;
                if (relu) v = fmaxf(v, 0.0f);
                Cb[(long long)mrow * ldc + col] = v;
            }
        }
    }
}

// ---------------------------------------------------------------------------
// LayerNorm (biased variance), one block of 256 per row of 768.
// ---------------------------------------------------------------------------
__global__ __launch_bounds__(256)
void layernorm_kernel(const float* __restrict__ in, const float* __restrict__ g,
                      const float* __restrict__ b, float* __restrict__ out)
{
    __shared__ float red[256];
    const int t = threadIdx.x;
    const float* xr = in + (long long)blockIdx.x * D_MODEL;

    float s = 0.f;
    for (int d = t; d < D_MODEL; d += 256) s += xr[d];
    red[t] = s; __syncthreads();
    for (int o = 128; o > 0; o >>= 1) { if (t < o) red[t] += red[t + o]; __syncthreads(); }
    const float mu = red[0] * (1.0f / D_MODEL);
    __syncthreads();

    float v = 0.f;
    for (int d = t; d < D_MODEL; d += 256) { float dx = xr[d] - mu; v += dx * dx; }
    red[t] = v; __syncthreads();
    for (int o = 128; o > 0; o >>= 1) { if (t < o) red[t] += red[t + o]; __syncthreads(); }
    const float rstd = rsqrtf(red[0] * (1.0f / D_MODEL) + LN_EPS);

    float* yr = out + (long long)blockIdx.x * D_MODEL;
    for (int d = t; d < D_MODEL; d += 256)
        yr[d] = (xr[d] - mu) * rstd * g[d] + b[d];
}

// ---------------------------------------------------------------------------
// Softmax over rows of length SEQ=128 (scores pre-scaled by dk^-0.5), in-place.
// ---------------------------------------------------------------------------
__global__ __launch_bounds__(128)
void softmax_rows128(float* __restrict__ att)
{
    __shared__ float red[128];
    const int t = threadIdx.x;
    float* r = att + (long long)blockIdx.x * SEQ;
    float val = r[t] * 0.125f;                      // 1/sqrt(64)
    red[t] = val; __syncthreads();
    for (int o = 64; o > 0; o >>= 1) { if (t < o) red[t] = fmaxf(red[t], red[t + o]); __syncthreads(); }
    const float mx = red[0]; __syncthreads();
    const float e = expf(val - mx);
    red[t] = e; __syncthreads();
    for (int o = 64; o > 0; o >>= 1) { if (t < o) red[t] += red[t + o]; __syncthreads(); }
    r[t] = e / red[0];
}

// ---------------------------------------------------------------------------
__global__ void embed_kernel(const int* __restrict__ ids, const float* __restrict__ table,
                             float* __restrict__ x)
{
    const int tkn = blockIdx.x;
    const long long src = (long long)ids[tkn] * D_MODEL;
    for (int d = threadIdx.x; d < D_MODEL; d += 256)
        x[(long long)tkn * D_MODEL + d] = table[src + d];
}

__global__ void add_kernel(float* __restrict__ x, const float* __restrict__ y, long long n)
{
    for (long long i = (long long)blockIdx.x * blockDim.x + threadIdx.x; i < n;
         i += (long long)gridDim.x * blockDim.x)
        x[i] += y[i];
}

// ---------------------------------------------------------------------------
// Router: logits = z @ w_sw + b_sw, softmax over 8 experts, argmax (first-max).
// ---------------------------------------------------------------------------
__global__ __launch_bounds__(256)
void router_kernel(const float* __restrict__ z, const float* __restrict__ w_sw,
                   const float* __restrict__ b_sw, float* __restrict__ rmax,
                   int* __restrict__ routes)
{
    const int t = blockIdx.x * blockDim.x + threadIdx.x;
    if (t >= T_TOK) return;
    const float* zr = z + (long long)t * D_MODEL;
    float logit[N_EXPERTS];
#pragma unroll
    for (int e = 0; e < N_EXPERTS; ++e) logit[e] = b_sw[e];
    for (int d = 0; d < D_MODEL; ++d) {
        const float zv = zr[d];
        const float* w = w_sw + d * N_EXPERTS;
#pragma unroll
        for (int e = 0; e < N_EXPERTS; ++e) logit[e] = fmaf(zv, w[e], logit[e]);
    }
    float mx = logit[0];
#pragma unroll
    for (int e = 1; e < N_EXPERTS; ++e) mx = fmaxf(mx, logit[e]);
    float p[N_EXPERTS]; float sum = 0.f;
#pragma unroll
    for (int e = 0; e < N_EXPERTS; ++e) { p[e] = expf(logit[e] - mx); sum += p[e]; }
    float best = -1.f; int arg = 0;
#pragma unroll
    for (int e = 0; e < N_EXPERTS; ++e) { const float pr = p[e] / sum; if (pr > best) { best = pr; arg = e; } }
    rmax[t] = best;
    routes[t] = arg;
}

// ---------------------------------------------------------------------------
// Deterministic in-order capacity scan (matches cumsum(onehot) semantics).
// One block, 256 threads x 16 tokens each.
// ---------------------------------------------------------------------------
__global__ __launch_bounds__(256)
void scan_kernel(const int* __restrict__ routes, int* __restrict__ pos, int* __restrict__ keep)
{
    __shared__ int cnt[256][N_EXPERTS];
    const int thr = threadIdx.x;
    const int t0  = thr * (T_TOK / 256);
    int c[N_EXPERTS];
#pragma unroll
    for (int e = 0; e < N_EXPERTS; ++e) c[e] = 0;
    for (int i = 0; i < T_TOK / 256; ++i) c[routes[t0 + i]]++;
#pragma unroll
    for (int e = 0; e < N_EXPERTS; ++e) cnt[thr][e] = c[e];
    __syncthreads();
    if (thr < N_EXPERTS) {                 // exclusive scan per expert over chunks
        int run = 0;
        for (int i = 0; i < 256; ++i) { const int tmp = cnt[i][thr]; cnt[i][thr] = run; run += tmp; }
    }
    __syncthreads();
#pragma unroll
    for (int e = 0; e < N_EXPERTS; ++e) c[e] = cnt[thr][e];
    for (int i = 0; i < T_TOK / 256; ++i) {
        const int t = t0 + i;
        const int e = routes[t];
        const int p = c[e]++;
        const int kp = (p < CAPACITY) ? 1 : 0;
        keep[t] = kp;
        pos[t]  = kp ? p : 0;
    }
}

__global__ __launch_bounds__(256)
void scatter_kernel(const float* __restrict__ z, const int* __restrict__ routes,
                    const int* __restrict__ pos, const int* __restrict__ keep,
                    float* __restrict__ buf)
{
    const int t = blockIdx.x;
    if (!keep[t]) return;
    const long long dst = ((long long)routes[t] * CAPACITY + pos[t]) * D_MODEL;
    const long long src = (long long)t * D_MODEL;
    for (int d = threadIdx.x; d < D_MODEL; d += 256) buf[dst + d] = z[src + d];
}

__global__ __launch_bounds__(256)
void gather_residual_kernel(float* __restrict__ x, const float* __restrict__ z,
                            const float* __restrict__ eo, const int* __restrict__ routes,
                            const int* __restrict__ pos, const int* __restrict__ keep,
                            const float* __restrict__ rmax)
{
    const int t = blockIdx.x;
    const float scale = rmax[t];
    const float* src = keep[t]
        ? eo + ((long long)routes[t] * CAPACITY + pos[t]) * D_MODEL
        : z + (long long)t * D_MODEL;
    float* xr = x + (long long)t * D_MODEL;
    for (int d = threadIdx.x; d < D_MODEL; d += 256)
        xr[d] += src[d] * scale;
}

// ---------------------------------------------------------------------------
extern "C" void kernel_launch(void* const* d_in, const int* in_sizes, int n_in,
                              void* d_out, int out_size, void* d_ws, size_t ws_size,
                              hipStream_t stream)
{
    (void)in_sizes; (void)n_in; (void)out_size; (void)ws_size;
    const int*   input_ids = (const int*)d_in[0];
    const float* embed     = (const float*)d_in[1];
    // params flattened in insertion order: 18 arrays per layer, then lnf_g/lnf_b
    auto LP = [&](int layer, int j) { return (const float*)d_in[2 + layer * 18 + j]; };
    const float* lnf_g = (const float*)d_in[2 + N_LAYERS * 18 + 0];
    const float* lnf_b = (const float*)d_in[2 + N_LAYERS * 18 + 1];

    const long long TD  = (long long)T_TOK * D_MODEL;               // 3,145,728
    const long long ATT = (long long)BATCH * HEADS * SEQ * SEQ;     // 6,291,456
    float* x        = (float*)d_ws;
    float* zb       = x + TD;
    float* q        = zb + TD;
    float* kb       = q + TD;
    float* vb       = kb + TD;
    float* att      = vb + TD;
    float* attn_out = att + ATT;
    float* h        = attn_out + TD;                                // 8*512*3072
    float* proj     = q;    // reuse (q dead after attention)
    float* buf      = kb;   // reuse
    float* eo       = vb;   // reuse
    float* rmax     = h + (long long)N_EXPERTS * CAPACITY * D_FF;
    int*   routes   = (int*)(rmax + T_TOK);
    int*   pos      = routes + T_TOK;
    int*   keep     = pos + T_TOK;

    embed_kernel<<<T_TOK, 256, 0, stream>>>(input_ids, embed, x);

    const dim3 gProj(D_MODEL / 128, T_TOK / 128, 1);                // 6 x 32
    const dim3 gQK(1, 1, BATCH * HEADS);                            // 128x128 tile
    const dim3 gAV(1, 1, BATCH * HEADS);                            // N=64 guarded
    const dim3 gF1(D_FF / 128, CAPACITY / 128, N_EXPERTS);          // 24 x 4 x 8
    const dim3 gF2(D_MODEL / 128, CAPACITY / 128, N_EXPERTS);       // 6 x 4 x 8
    const int BD = BATCH * D_MODEL;

    for (int L = 0; L < N_LAYERS; ++L) {
        const float *wq = LP(L, 0), *bq = LP(L, 1), *wk = LP(L, 2), *bk = LP(L, 3),
                    *wv = LP(L, 4), *bv = LP(L, 5), *wo = LP(L, 6), *bo = LP(L, 7),
                    *w_sw = LP(L, 8), *b_sw = LP(L, 9), *w1 = LP(L, 10), *b1 = LP(L, 11),
                    *w2 = LP(L, 12), *b2 = LP(L, 13), *ln1g = LP(L, 14), *ln1b = LP(L, 15),
                    *ln2g = LP(L, 16), *ln2b = LP(L, 17);

        // ---- attention block ------------------------------------------------
        layernorm_kernel<<<T_TOK, 256, 0, stream>>>(x, ln1g, ln1b, zb);

        gemm_bf16_wmma<<<gProj, 256, 0, stream>>>(zb, wq, bq, q, T_TOK, D_MODEL, D_MODEL,
            D_MODEL, D_MODEL, D_MODEL, 0, 0, 0, 0, 0, 0, 0, 0, 0);
        gemm_bf16_wmma<<<gProj, 256, 0, stream>>>(zb, wk, bk, kb, T_TOK, D_MODEL, D_MODEL,
            D_MODEL, D_MODEL, D_MODEL, 0, 0, 0, 0, 0, 0, 0, 0, 0);
        gemm_bf16_wmma<<<gProj, 256, 0, stream>>>(zb, wv, bv, vb, T_TOK, D_MODEL, D_MODEL,
            D_MODEL, D_MODEL, D_MODEL, 0, 0, 0, 0, 0, 0, 0, 0, 0);

        // scores[z=b*H+h][i][j] = Q_i . K_j  (B transposed: rows j, cols d)
        gemm_bf16_wmma<<<gQK, 256, 0, stream>>>(q, kb, nullptr, att, SEQ, SEQ, DK,
            BD, BD, SEQ, 0, 0, (long long)SEQ * SEQ, 0, 1, 1, 0, 1, 0);
        softmax_rows128<<<BATCH * HEADS * SEQ, 128, 0, stream>>>(att);
        // out = attn @ V
        gemm_bf16_wmma<<<gAV, 256, 0, stream>>>(att, vb, nullptr, attn_out, SEQ, DK, SEQ,
            SEQ, BD, BD, (long long)SEQ * SEQ, 0, 0, 0, 0, 1, 1, 0, 0);
        // output projection + residual
        gemm_bf16_wmma<<<gProj, 256, 0, stream>>>(attn_out, wo, bo, proj, T_TOK, D_MODEL, D_MODEL,
            D_MODEL, D_MODEL, D_MODEL, 0, 0, 0, 0, 0, 0, 0, 0, 0);
        add_kernel<<<4096, 256, 0, stream>>>(x, proj, TD);

        // ---- Switch-MoE block ----------------------------------------------
        layernorm_kernel<<<T_TOK, 256, 0, stream>>>(x, ln2g, ln2b, zb);
        router_kernel<<<T_TOK / 256, 256, 0, stream>>>(zb, w_sw, b_sw, rmax, routes);
        scan_kernel<<<1, 256, 0, stream>>>(routes, pos, keep);
        scatter_kernel<<<T_TOK, 256, 0, stream>>>(zb, routes, pos, keep, buf);

        gemm_bf16_wmma<<<gF1, 256, 0, stream>>>(buf, w1, b1, h, CAPACITY, D_FF, D_MODEL,
            D_MODEL, D_FF, D_FF,
            (long long)CAPACITY * D_MODEL, (long long)D_MODEL * D_FF,
            (long long)CAPACITY * D_FF, D_FF, 0, 0, 0, 0, 1 /*relu*/);
        gemm_bf16_wmma<<<gF2, 256, 0, stream>>>(h, w2, b2, eo, CAPACITY, D_MODEL, D_FF,
            D_FF, D_MODEL, D_MODEL,
            (long long)CAPACITY * D_FF, (long long)D_FF * D_MODEL,
            (long long)CAPACITY * D_MODEL, D_MODEL, 0, 0, 0, 0, 0);

        gather_residual_kernel<<<T_TOK, 256, 0, stream>>>(x, zb, eo, routes, pos, keep, rmax);
    }

    layernorm_kernel<<<T_TOK, 256, 0, stream>>>(x, lnf_g, lnf_b, (float*)d_out);
}
